// Coupling_Spline_NF_67242007986520
// MI455X (gfx1250) — matrix operand order repaired and checked by reading, not compiled, not run on hardware
//
#include <hip/hip_runtime.h>
#include <math.h>

// ---------------------------------------------------------------------------
// Coupling-spline normalizing flow log_prob for MI455X (gfx1250, wave32).
// Compute-bound on the 16->320->320->1008 MLPs (~56 GFLOP total) -> WMMA
// v_wmma_f32_16x16x32_f16 (f16 in, f32 accumulate). Spline math stays fp32.
// A fragments are held register-resident across N-tiles; both lane halves
// of each wave do spline work (2 output dims per iteration).
// ---------------------------------------------------------------------------

typedef _Float16 v16h __attribute__((ext_vector_type(16)));
typedef _Float16 v8h  __attribute__((ext_vector_type(8)));
typedef float    v8f  __attribute__((ext_vector_type(8)));

#define NROWS   32768
#define DFULL   32
#define DHALF   16
#define HID     320
#define POUTP   1024      // reordered/padded W3 column count (16 dims * 64)

#define F_MBW   0.001f
#define F_MBH   0.001f
#define F_MD    0.001f
#define F_ML    0.025f
#define F_LOG2PI 1.8378770664093453f

// weight tiling (B fragments): Kpad multiple of 32, N multiple of 16
#define KT_W1 1
#define NT_W1 20
#define KT_W2 10
#define NT_W2 20
#define KT_W3 10
#define NT_W3 64

// spline table layout (floats, per transform)
#define TBL_WID   0        // [16][16]
#define TBL_CUMW  256      // [16][17]
#define TBL_HGT   528      // [16][16]
#define TBL_CUMH  784      // [16][17]
#define TBL_DPAD  1056     // [16][17]
#define TBL_LAM   1328     // [16][16]
#define TBL_FLOATS 1584
#define TBL_BYTES  6336    // 1584*4, multiple of 64

// per-wave LDS region
#define W_OFF_PTILE 0          // float [2][16][64]  8192 (two dims at once)
#define W_OFF_X1F   8192       // float [16][16]     1024
#define W_OFF_Y2F   9216       // float [16][16]     1024
#define W_OFF_X2F   10240      // float [16][16]     1024
#define W_OFF_LACC  11264      // float [32]          128
#define W_OFF_ABUF  11392      // half  [16][32]     1024
#define W_OFF_H1    12416      // half  [16][320]   10240
#define W_OFF_H2    22656      // half  [16][320]   10240
#define WAVE_BYTES  32896
#define SMEM_BYTES  (TBL_BYTES + 4 * WAVE_BYTES)   // 137920 (< 320KB/WGP)

// ---------------------------------------------------------------------------
// device helpers
// ---------------------------------------------------------------------------
__device__ __forceinline__ float softplusf(float x) {
    return (x > 20.f) ? x : log1pf(expf(x));
}

__device__ __forceinline__ v8f wmma_f16(v16h a, v16h b, v8f c) {
    // emits v_wmma_f32_16x16x32_f16
    return __builtin_amdgcn_wmma_f32_16x16x32_f16(false, a, false, b,
                                                  (short)0, c, false, false);
}

// A fragment (16x32 f16, wave32, contiguous k order as in the B packing): lane L -> row L&15, half L>>4.
// element e : K = k0 + 16*half + e
__device__ __forceinline__ v16h load_a_frag(const _Float16* h, int lane,
                                            int k0, int stride) {
    int row = lane & 15, hf = lane >> 4;
    const v8h* p0 = (const v8h*)(h + row * stride + k0 + 16 * hf);
    const v8h* p1 = (const v8h*)(h + row * stride + k0 + 16 * hf + 8);
    v8h lo = *p0, hi = *p1;
    v16h a;
#pragma unroll
    for (int e = 0; e < 8; ++e) { a[e] = lo[e]; a[8 + e] = hi[e]; }
    return a;
}

// rational-linear spline inverse core (per element)
__device__ __forceinline__ void rls_core(float y, float yc, bool inside,
                                         float iw, float icw, float ih,
                                         float ich, float idv, float idvp1,
                                         float il, float& xo, float& lo_) {
    float idel = ih / iw;
    float wb = sqrtf(idv / idvp1);
    float wc = (il * idv + (1.f - il) * wb * idvp1) / idel;
    float ya = ich;
    float yb = ih + ich;
    float ym = ((1.f - il) * ya + il * wb * yb) / (1.f - il + il * wb);
    bool left = (yc <= ym);
    float num = left ? il * (ya - yc)
                     : (wc - il * wb) * yc + il * wb * yb - wc * ym;
    float den = left ? (wc - 1.f) * yc + ya - wc * ym
                     : (wc - wb) * yc + wb * yb - wc * ym;
    float x = num / den * iw + icw;
    float dnum = (left ? wc * il * (ym - ya)
                       : wb * wc * (1.f - il) * (yb - ym)) * iw;
    float ladj = logf(dnum) - 2.f * logf(fabsf(den));
    xo  = inside ? x : y;
    lo_ = inside ? ladj : 0.f;
}

// stage 1: batch-constant tables (in LDS)
__device__ __forceinline__ void spline_stage1(const float* tbl, int d, float y,
                                              float& xo, float& lo_) {
    bool inside = (y >= -3.f) && (y <= 3.f);
    float yc = fminf(fmaxf(y, -3.f), 3.f);
    int cnt = 0;
#pragma unroll
    for (int k = 0; k < 17; ++k)
        cnt += (yc >= tbl[TBL_CUMH + d * 17 + k]) ? 1 : 0;
    int idx = cnt - 1; idx = idx < 0 ? 0 : (idx > 15 ? 15 : idx);
    float iw    = tbl[TBL_WID  + d * 16 + idx];
    float icw   = tbl[TBL_CUMW + d * 17 + idx];
    float ih    = tbl[TBL_HGT  + d * 16 + idx];
    float ich   = tbl[TBL_CUMH + d * 17 + idx];
    float idv   = tbl[TBL_DPAD + d * 17 + idx];
    float idvp1 = tbl[TBL_DPAD + d * 17 + idx + 1];
    float il    = tbl[TBL_LAM  + d * 16 + idx];
    rls_core(y, yc, inside, iw, icw, ih, ich, idv, idvp1, il, xo, lo_);
}

// stage 2: per-sample raw params p[63] in LDS:
// [0..15]=w_raw [16..31]=h_raw [32..46]=d_raw [47..62]=l_raw
__device__ __forceinline__ void spline_stage2(const float* p, float y,
                                              float& xo, float& lo_) {
    bool inside = (y >= -3.f) && (y <= 3.f);
    float yc = fminf(fmaxf(y, -3.f), 3.f);

    // softmax maxima / sums for w_raw and h_raw
    float hmax = -1e30f, wmax = -1e30f;
#pragma unroll
    for (int b = 0; b < 16; ++b) {
        hmax = fmaxf(hmax, p[16 + b]);
        wmax = fmaxf(wmax, p[b]);
    }
    float hsum = 0.f, wsum = 0.f;
#pragma unroll
    for (int b = 0; b < 16; ++b) {
        hsum += expf(p[16 + b] - hmax);
        wsum += expf(p[b] - wmax);
    }
    float hscale = (1.f - 16.f * F_MBH) / hsum;
    float wscale = (1.f - 16.f * F_MBW) / wsum;

    // branchless bin search on height knots
    int idx = 15; float ich = -3.f, ih = 6.f;
    float ph = 0.f; bool found = false;
#pragma unroll
    for (int b = 0; b < 16; ++b) {
        float hb = F_MBH + hscale * expf(p[16 + b] - hmax);
        float lo = (b == 0) ? -3.f : fmaf(6.f, ph, -3.f);
        ph += hb;
        float hi = (b == 15) ? 3.f : fmaf(6.f, ph, -3.f);
        bool take = (!found) && ((yc < hi) || (b == 15));
        if (take) { idx = b; ich = lo; ih = hi - lo; found = true; }
    }
    float pw = 0.f, icw = -3.f, iw = 6.f;
#pragma unroll
    for (int b = 0; b < 16; ++b) {
        float wb_ = F_MBW + wscale * expf(p[b] - wmax);
        float lo = (b == 0) ? -3.f : fmaf(6.f, pw, -3.f);
        pw += wb_;
        float hi = (b == 15) ? 3.f : fmaf(6.f, pw, -3.f);
        if (b == idx) { icw = lo; iw = hi - lo; }
    }
    float idv   = (idx == 0)  ? (1.f - F_MD) : (F_MD + softplusf(p[32 + idx - 1]));
    float idvp1 = (idx == 15) ? (1.f - F_MD) : (F_MD + softplusf(p[32 + idx]));
    float il = F_ML + (1.f - 2.f * F_ML) / (1.f + expf(-p[47 + idx]));
    rls_core(y, yc, inside, iw, icw, ih, ich, idv, idvp1, il, xo, lo_);
}

// ---------------------------------------------------------------------------
// prep kernels
// ---------------------------------------------------------------------------

// pack row-major f32 weight [Kreal x Ncols] into f16 WMMA-B fragment tiles:
// tile t = nt*numKT + kt ; lane L ; element e -> out[(t*32+L)*16 + e]
// value = W[K][col], K = kt*32 + 16*(L>>4) + e, col = nt*16 + (L&15)
__global__ void pack_plain_kernel(const float* __restrict__ W,
                                  _Float16* __restrict__ out,
                                  int Kreal, int Ncols, int numKT, int numNT) {
    int total = numKT * numNT * 32 * 16;
    for (int i = blockIdx.x * blockDim.x + threadIdx.x; i < total;
         i += gridDim.x * blockDim.x) {
        int e = i & 15;
        int L = (i >> 4) & 31;
        int t = i >> 9;
        int kt = t % numKT;
        int nt = t / numKT;
        int col = nt * 16 + (L & 15);
        int K = kt * 32 + 16 * (L >> 4) + e;
        float v = (K < Kreal) ? W[(size_t)K * Ncols + col] : 0.f;
        out[i] = (_Float16)v;
    }
}

// column permutation for W3: group the 63 params of each output dim j into one
// contiguous 64-column block: [w(16) h(16) d(15) l(16) pad(1)]
__device__ __forceinline__ int w3_orig_col(int colp) {
    int j = colp >> 6, c = colp & 63;
    if (c < 16) return 16 * j + c;               // w_raw
    if (c < 32) return 256 + 16 * j + (c - 16);  // h_raw
    if (c < 47) return 512 + 15 * j + (c - 32);  // d_raw
    if (c < 63) return 752 + 16 * j + (c - 47);  // l_raw
    return -1;                                   // pad
}

__global__ void pack_w3_kernel(const float* __restrict__ W3,
                               _Float16* __restrict__ out) {
    int total = KT_W3 * NT_W3 * 32 * 16;
    for (int i = blockIdx.x * blockDim.x + threadIdx.x; i < total;
         i += gridDim.x * blockDim.x) {
        int e = i & 15;
        int L = (i >> 4) & 31;
        int t = i >> 9;
        int kt = t % KT_W3;
        int nt = t / KT_W3;
        int colp = nt * 16 + (L & 15);
        int K = kt * 32 + 16 * (L >> 4) + e;
        int orig = w3_orig_col(colp);
        float v = (orig >= 0 && K < HID) ? W3[(size_t)K * 1008 + orig] : 0.f;
        out[i] = (_Float16)v;
    }
}

__global__ void reorder_b3_kernel(const float* __restrict__ b3,
                                  float* __restrict__ b3p) {
    for (int colp = blockIdx.x * blockDim.x + threadIdx.x; colp < POUTP;
         colp += gridDim.x * blockDim.x) {
        int orig = w3_orig_col(colp);
        b3p[colp] = (orig >= 0) ? b3[orig] : 0.f;
    }
}

// batch-constant spline tables for one transform (one thread per dim)
__global__ void spline_tables_kernel(const float* __restrict__ w_raw,
                                     const float* __restrict__ h_raw,
                                     const float* __restrict__ d_raw,
                                     const float* __restrict__ l_raw,
                                     float* __restrict__ tbl) {
    int d = threadIdx.x;
    if (d >= 16) return;

    // widths + cumw
    {
        float mx = -1e30f;
        for (int b = 0; b < 16; ++b) mx = fmaxf(mx, w_raw[d * 16 + b]);
        float s = 0.f;
        for (int b = 0; b < 16; ++b) s += expf(w_raw[d * 16 + b] - mx);
        float sc = (1.f - 16.f * F_MBW) / s;
        float cw[17]; cw[0] = -3.f;
        float acc = 0.f;
        for (int b = 0; b < 16; ++b) {
            acc += F_MBW + sc * expf(w_raw[d * 16 + b] - mx);
            cw[b + 1] = (b == 15) ? 3.f : fmaf(6.f, acc, -3.f);
        }
        for (int b = 0; b < 16; ++b) {
            tbl[TBL_WID + d * 16 + b] = cw[b + 1] - cw[b];
            tbl[TBL_CUMW + d * 17 + b] = cw[b];
        }
        tbl[TBL_CUMW + d * 17 + 16] = cw[16];
    }
    // heights + cumh
    {
        float mx = -1e30f;
        for (int b = 0; b < 16; ++b) mx = fmaxf(mx, h_raw[d * 16 + b]);
        float s = 0.f;
        for (int b = 0; b < 16; ++b) s += expf(h_raw[d * 16 + b] - mx);
        float sc = (1.f - 16.f * F_MBH) / s;
        float ch[17]; ch[0] = -3.f;
        float acc = 0.f;
        for (int b = 0; b < 16; ++b) {
            acc += F_MBH + sc * expf(h_raw[d * 16 + b] - mx);
            ch[b + 1] = (b == 15) ? 3.f : fmaf(6.f, acc, -3.f);
        }
        for (int b = 0; b < 16; ++b) {
            tbl[TBL_HGT + d * 16 + b] = ch[b + 1] - ch[b];
            tbl[TBL_CUMH + d * 17 + b] = ch[b];
        }
        tbl[TBL_CUMH + d * 17 + 16] = ch[16];
    }
    // padded derivatives + lambda
    tbl[TBL_DPAD + d * 17 + 0] = 1.f - F_MD;
    for (int k = 1; k <= 15; ++k)
        tbl[TBL_DPAD + d * 17 + k] = F_MD + softplusf(d_raw[d * 15 + (k - 1)]);
    tbl[TBL_DPAD + d * 17 + 16] = 1.f - F_MD;
    for (int b = 0; b < 16; ++b)
        tbl[TBL_LAM + d * 16 + b] =
            F_ML + (1.f - 2.f * F_ML) / (1.f + expf(-l_raw[d * 16 + b]));
}

__global__ void lpaff_kernel(const float* __restrict__ stds,
                             float* __restrict__ out) {
    if (threadIdx.x == 0 && blockIdx.x == 0) {
        float s = 0.f;
        for (int c = 0; c < DFULL; ++c) s += logf(10.f * stds[c]);
        *out = -s;
    }
}

// ---------------------------------------------------------------------------
// main flow-stage kernel: 128 threads = 4 wave32, 16 rows per wave
// ---------------------------------------------------------------------------
__global__ __launch_bounds__(128)
void flow_stage_kernel(const float* __restrict__ in,
                       const float* __restrict__ means,
                       const float* __restrict__ stds,
                       const _Float16* __restrict__ W1p,
                       const _Float16* __restrict__ W2p,
                       const _Float16* __restrict__ W3p,
                       const float* __restrict__ b1,
                       const float* __restrict__ b2,
                       const float* __restrict__ b3p,
                       const float* __restrict__ tblg,
                       float* __restrict__ zout,
                       float* __restrict__ ladj,
                       float* __restrict__ out,
                       const float* __restrict__ lp_aff,
                       int do_affine, int do_final) {
    extern __shared__ char smem[];
    const int lane = threadIdx.x & 31;
    const int wave = threadIdx.x >> 5;
    const int row0 = (blockIdx.x * 4 + wave) * 16;

    float* tbl = (float*)smem;
    char* wb = smem + TBL_BYTES + (size_t)wave * WAVE_BYTES;
    float*    ptile   = (float*)(wb + W_OFF_PTILE);   // [2][16][64]
    float*    x1f     = (float*)(wb + W_OFF_X1F);
    float*    y2f     = (float*)(wb + W_OFF_Y2F);
    float*    x2f     = (float*)(wb + W_OFF_X2F);
    float*    laneacc = (float*)(wb + W_OFF_LACC);    // [32]
    _Float16* Abuf    = (_Float16*)(wb + W_OFF_ABUF);
    _Float16* h1      = (_Float16*)(wb + W_OFF_H1);
    _Float16* h2      = (_Float16*)(wb + W_OFF_H2);

    float accl = 0.f;   // per-lane log-det accumulator

    // 0) tables -> LDS (block cooperative)
    for (int i = threadIdx.x; i < TBL_FLOATS; i += blockDim.x) tbl[i] = tblg[i];
    __syncthreads();

    // 1) load 16 rows x 32 cols (+ optional affine), split halves
    {
        int c = lane;
        float scale = 1.f, mean = 0.f;
        if (do_affine) { mean = means[c]; scale = 1.f / (10.f * stds[c]); }
        for (int r = 0; r < 16; ++r) {
            float v = in[(size_t)(row0 + r) * DFULL + c];
            if (do_affine) v = (v - mean) * scale;
            if (c < 16) x1f[r * 16 + c] = v;
            else        y2f[r * 16 + (c - 16)] = v;
        }
    }
    __syncthreads();

    // 2) stage-1 spline (const tables): all 32 lanes, 8 dims per lane half
    {
        int row = lane & 15;
        int d0 = (lane >> 4) * 8;
#pragma unroll
        for (int dd = 0; dd < 8; ++dd) {
            int d = d0 + dd;
            float x, la;
            spline_stage1(tbl, d, x1f[row * 16 + d], x, la);
            x1f[row * 16 + d] = x;
            Abuf[row * 32 + d] = (_Float16)x;
            Abuf[row * 32 + 16 + d] = (_Float16)0.f;
            accl += la;
        }
    }
    __syncthreads();

    const int coll = lane & 15, hf = lane >> 4;

    // 3) GEMM1: h1 = relu(x1 @ W1 + b1)   [16x16 @ 16x320]
    {
        v16h a0 = load_a_frag(Abuf, lane, 0, 32);
        for (int nt = 0; nt < NT_W1; ++nt) {
            v16h b = *(const v16h*)(W1p + (size_t)(nt * 32 + lane) * 16);
            v8f c = {};
            c = wmma_f16(a0, b, c);
            int col = nt * 16 + coll;
            float bias = b1[col];
#pragma unroll
            for (int v = 0; v < 8; ++v) {
                int row = v + 8 * hf;
                h1[row * HID + col] = (_Float16)fmaxf(c[v] + bias, 0.f);
            }
        }
    }
    __syncthreads();

    // 4) GEMM2: h2 = relu(h1 @ W2 + b2)   [16x320 @ 320x320]
    {
        v16h afr[KT_W2];                 // A fragments held in registers
#pragma unroll
        for (int kt = 0; kt < KT_W2; ++kt)
            afr[kt] = load_a_frag(h1, lane, kt * 32, HID);
        for (int nt = 0; nt < NT_W2; ++nt) {
            if (nt + 1 < NT_W2)
                __builtin_prefetch(W2p + (size_t)((nt + 1) * KT_W2 * 32) * 16, 0, 1);
            v8f c = {};
#pragma unroll
            for (int kt = 0; kt < KT_W2; ++kt) {
                v16h b = *(const v16h*)(W2p + (size_t)((nt * KT_W2 + kt) * 32 + lane) * 16);
                c = wmma_f16(afr[kt], b, c);
            }
            int col = nt * 16 + coll;
            float bias = b2[col];
#pragma unroll
            for (int v = 0; v < 8; ++v) {
                int row = v + 8 * hf;
                h2[row * HID + col] = (_Float16)fmaxf(c[v] + bias, 0.f);
            }
        }
    }
    __syncthreads();

    // 5) GEMM3 + stage-2 spline, two output dims per iteration.
    //    dims (2jj, 2jj+1) -> 8 column tiles nt = jj*8 + ct
    {
        v16h afr[KT_W3];                 // h2 A fragments, register resident
#pragma unroll
        for (int kt = 0; kt < KT_W3; ++kt)
            afr[kt] = load_a_frag(h2, lane, kt * 32, HID);

        for (int jj = 0; jj < 8; ++jj) {
            for (int ct = 0; ct < 8; ++ct) {
                int nt = jj * 8 + ct;
                if (nt + 1 < NT_W3)
                    __builtin_prefetch(W3p + (size_t)((nt + 1) * KT_W3 * 32) * 16, 0, 1);
                v8f c = {};
#pragma unroll
                for (int kt = 0; kt < KT_W3; ++kt) {
                    v16h b = *(const v16h*)(W3p + (size_t)((nt * KT_W3 + kt) * 32 + lane) * 16);
                    c = wmma_f16(afr[kt], b, c);
                }
                float bias = b3p[nt * 16 + coll];
                float* pt = ptile + (ct >> 2) * 1024 + (ct & 3) * 16 + coll;
#pragma unroll
                for (int v = 0; v < 8; ++v)
                    pt[(v + 8 * hf) * 64] = c[v] + bias;
            }
            __syncthreads();
            {
                int row = lane & 15;
                int half = lane >> 4;
                int j = 2 * jj + half;                 // dim handled by this lane
                const float* p = ptile + half * 1024 + row * 64;
                float x, la;
                spline_stage2(p, y2f[row * 16 + j], x, la);
                x2f[row * 16 + j] = x;
                accl += la;
            }
            __syncthreads();
        }
    }

    // combine per-lane log-det halves
    laneacc[lane] = accl;
    __syncthreads();

    // 6) epilogue
    if (!do_final) {
        int c = lane;
        for (int r = 0; r < 16; ++r) {
            float v = (c < 16) ? x1f[r * 16 + c] : x2f[r * 16 + (c - 16)];
            zout[(size_t)(row0 + r) * DFULL + c] = v;
        }
        if (lane < 16) ladj[row0 + lane] = laneacc[lane] + laneacc[lane + 16];
    } else {
        if (lane < 16) {
            int row = lane;
            float s = 0.f;
#pragma unroll
            for (int d = 0; d < 16; ++d) {
                float a = x1f[row * 16 + d], b = x2f[row * 16 + d];
                s += a * a + b * b;
            }
            float base = -0.5f * s - 0.5f * (float)DFULL * F_LOG2PI;
            float rtot = laneacc[row] + laneacc[row + 16];
            out[row0 + row] = base + rtot + ladj[row0 + row] + lp_aff[0];
        }
    }
}

// ---------------------------------------------------------------------------
// host launcher
// ---------------------------------------------------------------------------
extern "C" void kernel_launch(void* const* d_in, const int* in_sizes, int n_in,
                              void* d_out, int out_size, void* d_ws, size_t ws_size,
                              hipStream_t stream) {
    (void)in_sizes; (void)n_in; (void)out_size; (void)ws_size;

    const float* data  = (const float*)d_in[0];
    const float* means = (const float*)d_in[1];
    const float* stds  = (const float*)d_in[2];
    // t index 0 = t1 (inputs 3..12), 1 = t2 (inputs 13..22)
    const float* W_[2]  = { (const float*)d_in[3],  (const float*)d_in[13] };
    const float* H_[2]  = { (const float*)d_in[4],  (const float*)d_in[14] };
    const float* D_[2]  = { (const float*)d_in[5],  (const float*)d_in[15] };
    const float* L_[2]  = { (const float*)d_in[6],  (const float*)d_in[16] };
    const float* W1[2]  = { (const float*)d_in[7],  (const float*)d_in[17] };
    const float* B1[2]  = { (const float*)d_in[8],  (const float*)d_in[18] };
    const float* W2[2]  = { (const float*)d_in[9],  (const float*)d_in[19] };
    const float* B2[2]  = { (const float*)d_in[10], (const float*)d_in[20] };
    const float* W3[2]  = { (const float*)d_in[11], (const float*)d_in[21] };
    const float* B3[2]  = { (const float*)d_in[12], (const float*)d_in[22] };

    char* ws = (char*)d_ws;
    size_t off = 0;
    auto take = [&](size_t bytes) -> size_t {
        off = (off + 255) & ~(size_t)255;
        size_t o = off; off += bytes; return o;
    };
    size_t o_lpaff = take(4);
    size_t o_zbuf  = take((size_t)NROWS * DFULL * 4);
    size_t o_ladj  = take((size_t)NROWS * 4);
    size_t o_tbl[2], o_b3p[2], o_w1p[2], o_w2p[2], o_w3p[2];
    for (int t = 0; t < 2; ++t) {
        o_tbl[t] = take(TBL_FLOATS * 4);
        o_b3p[t] = take(POUTP * 4);
        o_w1p[t] = take((size_t)KT_W1 * NT_W1 * 512 * 2);
        o_w2p[t] = take((size_t)KT_W2 * NT_W2 * 512 * 2);
        o_w3p[t] = take((size_t)KT_W3 * NT_W3 * 512 * 2);
    }

    float* lpaff = (float*)(ws + o_lpaff);
    float* zbuf  = (float*)(ws + o_zbuf);
    float* ladjb = (float*)(ws + o_ladj);

    for (int t = 0; t < 2; ++t) {
        _Float16* w1p = (_Float16*)(ws + o_w1p[t]);
        _Float16* w2p = (_Float16*)(ws + o_w2p[t]);
        _Float16* w3p = (_Float16*)(ws + o_w3p[t]);
        float* b3p = (float*)(ws + o_b3p[t]);
        float* tbl = (float*)(ws + o_tbl[t]);

        int tot1 = KT_W1 * NT_W1 * 512;
        int tot2 = KT_W2 * NT_W2 * 512;
        int tot3 = KT_W3 * NT_W3 * 512;
        pack_plain_kernel<<<(tot1 + 255) / 256, 256, 0, stream>>>(
            W1[t], w1p, DHALF, HID, KT_W1, NT_W1);
        pack_plain_kernel<<<(tot2 + 255) / 256, 256, 0, stream>>>(
            W2[t], w2p, HID, HID, KT_W2, NT_W2);
        pack_w3_kernel<<<(tot3 + 255) / 256, 256, 0, stream>>>(W3[t], w3p);
        reorder_b3_kernel<<<4, 256, 0, stream>>>(B3[t], b3p);
        spline_tables_kernel<<<1, 32, 0, stream>>>(W_[t], H_[t], D_[t], L_[t], tbl);
    }
    lpaff_kernel<<<1, 1, 0, stream>>>(stds, lpaff);

    dim3 grid(NROWS / 64), block(128);
    // transform t2 first: affine input, write z1 + a2
    flow_stage_kernel<<<grid, block, SMEM_BYTES, stream>>>(
        data, means, stds,
        (const _Float16*)(ws + o_w1p[1]), (const _Float16*)(ws + o_w2p[1]),
        (const _Float16*)(ws + o_w3p[1]),
        B1[1], B2[1], (const float*)(ws + o_b3p[1]),
        (const float*)(ws + o_tbl[1]),
        zbuf, ladjb, (float*)d_out, lpaff,
        /*do_affine=*/1, /*do_final=*/0);
    // transform t1: consume z1, emit final log-prob
    flow_stage_kernel<<<grid, block, SMEM_BYTES, stream>>>(
        zbuf, means, stds,
        (const _Float16*)(ws + o_w1p[0]), (const _Float16*)(ws + o_w2p[0]),
        (const _Float16*)(ws + o_w3p[0]),
        B1[0], B2[0], (const float*)(ws + o_b3p[0]),
        (const float*)(ws + o_tbl[0]),
        zbuf, ladjb, (float*)d_out, lpaff,
        /*do_affine=*/0, /*do_final=*/1);
}
